// RecognitionLattice_13426067767732
// MI455X (gfx1250) — compile-verified, matrix operand order checked by reading, not run on hardware
//
#include <hip/hip_runtime.h>
#include <hip/hip_bf16.h>

#define Bd   4
#define Td   512
#define Ud   96
#define Fd   512
#define Hd   512
#define Vp1  257
#define UPAD 112
#define NPW  320     // padded N for WoT rows (20 tiles of 16)
#define NEGINF -1e30f

typedef __attribute__((ext_vector_type(16))) __bf16        v16bf;
typedef __attribute__((ext_vector_type(8)))  float         v8f;
typedef __attribute__((ext_vector_type(4)))  unsigned int  u32x4;

union BFrag { v16bf v; u32x4 q[2]; };

__device__ __forceinline__ unsigned short f2bf(float f) {
  union { float f; unsigned int u; } x; x.f = f;
  unsigned int u = x.u;
  unsigned int r = u + 0x7FFFu + ((u >> 16) & 1u);  // RNE
  return (unsigned short)(r >> 16);
}

// ---------------- prep kernels ----------------
__global__ void prep_frames_bf(const float* __restrict__ fr, unsigned short* __restrict__ out) {
  int i = blockIdx.x * blockDim.x + threadIdx.x;
  if (i < Bd * Td * Fd) out[i] = f2bf(fr[i]);
}

__global__ void prep_wfT(const float* __restrict__ Wf, unsigned short* __restrict__ out) {
  int i = blockIdx.x * blockDim.x + threadIdx.x;
  if (i < Hd * Fd) {
    int h = i >> 9, f = i & 511;                 // out[h][f] = Wf[f][h]
    out[i] = f2bf(Wf[f * Hd + h]);
  }
}

__global__ void prep_woT(const float* __restrict__ Wo, unsigned short* __restrict__ out) {
  int i = blockIdx.x * blockDim.x + threadIdx.x;
  if (i < NPW * Hd) {
    int n = i >> 9, h = i & 511;                 // out[n][h] = Wo[h][n], zero-pad n>=257
    float v = (n < Vp1) ? Wo[h * Vp1 + n] : 0.0f;
    out[i] = f2bf(v);
  }
}

__global__ void prep_cemb(const float* __restrict__ E, const int* __restrict__ labels,
                          float* __restrict__ out) {
  int i = blockIdx.x * blockDim.x + threadIdx.x;
  if (i < Bd * UPAD * Hd) {
    int b = i / (UPAD * Hd);
    int r = i - b * (UPAD * Hd);
    int u = r >> 9, h = r & 511;
    float v;
    if (u > Ud) v = 0.0f;                        // padded rows
    else {
      int c = (u == 0) ? 0 : labels[b * Ud + (u - 1)];  // ctx state
      v = E[c * Hd + h];
    }
    out[i] = v;
  }
}

// ---------------- fproj = frames @ Wf (bf16 WMMA, f32 acc) ----------------
__global__ __launch_bounds__(128) void fproj_kernel(
    const unsigned short* __restrict__ frames_bf,  // [2048][512]
    const unsigned short* __restrict__ wfT,        // [512][512]  (h-major, f contiguous)
    float* __restrict__ fproj)                     // [2048][512]
{
  int wave = threadIdx.x >> 5;
  int lane = threadIdx.x & 31;
  int tile = blockIdx.x * 4 + wave;              // 4096 tiles: 128 m-tiles x 32 n-tiles
  int mt = tile >> 5, nt = tile & 31;
  int mrow = lane & 15, half = lane >> 4;

  const unsigned short* arow = frames_bf + (size_t)(mt * 16 + mrow) * Fd;
  const unsigned short* brow = wfT       + (size_t)(nt * 16 + mrow) * Fd;

  v8f acc = {};
#pragma unroll
  for (int kt = 0; kt < 16; ++kt) {
    int kb1 = kt * 32 + half * 8;
    int kb2 = kb1 + 16;
    BFrag a, bfr;
    a.q[0]   = *(const u32x4*)&arow[kb1];
    a.q[1]   = *(const u32x4*)&arow[kb2];
    bfr.q[0] = *(const u32x4*)&brow[kb1];
    bfr.q[1] = *(const u32x4*)&brow[kb2];
    acc = __builtin_amdgcn_wmma_f32_16x16x32_bf16(false, a.v, false, bfr.v,
                                                  (short)0, acc, false, false);
  }
  float* out = fproj + (size_t)(mt * 16) * Hd + nt * 16;
#pragma unroll
  for (int r = 0; r < 8; ++r)
    out[(size_t)(r + 8 * half) * Hd + mrow] = acc[r];
}

// ------ fused joint: tanh + GEMM(Wo) + log-softmax -> blank/lex ------
// 2 t-values per block: each Wo fragment in registers feeds 2 WMMAs
// (halves the dominant L2 traffic; B fragments are t-invariant).
__global__ __launch_bounds__(128) void joint_kernel(
    const float* __restrict__ fproj,          // [B][T][512]
    const float* __restrict__ cemb,           // [B][112][512]
    const unsigned short* __restrict__ wot,   // [320][512] bf16
    const int* __restrict__ labels,           // [B][96]
    float* __restrict__ blank_out,            // [B][T][112]
    float* __restrict__ lex_out)              // [B][T][112]
{
  __shared__ __align__(16) unsigned short Atile[2][16][520];  // bf16, row-padded
  __shared__ float Lg[16][324];                               // logits tile, row-padded
  const int blk = blockIdx.x;
  const int ub = blk % 7;
  const int tp = (blk / 7) % (Td / 2);
  const int b  = blk / (7 * (Td / 2));
  const int u0 = ub * 16;
  const int t0 = tp * 2;

  const float* ce = cemb + ((size_t)b * UPAD + u0) * Hd;
  const int tid = threadIdx.x;

  // Build both A tiles: A[tt][m][k] = tanh(fproj[t0+tt][k] + cemb[u0+m][k])
  for (int i = tid; i < 2 * 16 * Hd; i += 128) {
    int tt = i >> 13;
    int rem = i & 8191;
    int m = rem >> 9, k = rem & 511;
    const float* fp = fproj + ((size_t)(b * Td + t0 + tt)) * Hd;
    Atile[tt][m][k] = f2bf(tanhf(fp[k] + ce[m * Hd + k]));
  }
  __syncthreads();

  const int lane = tid & 31;
  const int wave = tid >> 5;
  const int mrow = lane & 15;
  const int half = lane >> 4;

  v8f acc[2][5] = {};
#pragma unroll 2
  for (int kt = 0; kt < 16; ++kt) {
    int kb1 = kt * 32 + half * 8;
    int kb2 = kb1 + 16;
    BFrag a0, a1;
    a0.q[0] = *(const u32x4*)&Atile[0][mrow][kb1];
    a0.q[1] = *(const u32x4*)&Atile[0][mrow][kb2];
    a1.q[0] = *(const u32x4*)&Atile[1][mrow][kb1];
    a1.q[1] = *(const u32x4*)&Atile[1][mrow][kb2];
#pragma unroll
    for (int j = 0; j < 5; ++j) {
      int n = (wave + 4 * j) * 16 + mrow;            // n < 320
      const unsigned short* wp = wot + (size_t)n * Hd;
      BFrag bfr;
      bfr.q[0] = *(const u32x4*)&wp[kb1];
      bfr.q[1] = *(const u32x4*)&wp[kb2];
      acc[0][j] = __builtin_amdgcn_wmma_f32_16x16x32_bf16(false, a0.v, false, bfr.v,
                                                          (short)0, acc[0][j], false, false);
      acc[1][j] = __builtin_amdgcn_wmma_f32_16x16x32_bf16(false, a1.v, false, bfr.v,
                                                          (short)0, acc[1][j], false, false);
    }
  }

#pragma unroll
  for (int tt = 0; tt < 2; ++tt) {
    // Dump C tiles for this t to LDS (disjoint columns per wave)
#pragma unroll
    for (int j = 0; j < 5; ++j) {
      int n = (wave + 4 * j) * 16 + mrow;
#pragma unroll
      for (int r = 0; r < 8; ++r)
        Lg[r + 8 * half][n] = acc[tt][j][r];
    }
    __syncthreads();

    // log-sum-exp over the 257 valid columns; 8 threads per row
    const int r  = tid >> 3;
    const int l8 = tid & 7;
    float mx = -3.0e38f;
    for (int c = l8; c < Vp1; c += 8) mx = fmaxf(mx, Lg[r][c]);
#pragma unroll
    for (int s = 1; s < 8; s <<= 1) mx = fmaxf(mx, __shfl_xor(mx, s, 32));
    float sum = 0.0f;
    for (int c = l8; c < Vp1; c += 8) sum += __expf(Lg[r][c] - mx);
#pragma unroll
    for (int s = 1; s < 8; s <<= 1) sum += __shfl_xor(sum, s, 32);
    float lse = mx + __logf(sum);

    if (l8 == 0) {
      int u = u0 + r;
      if (u <= Ud) {
        size_t base = ((size_t)(b * Td + (t0 + tt))) * UPAD + u;
        blank_out[base] = Lg[r][0] - lse;
        if (u < Ud) {
          int lbl = labels[b * Ud + u];            // 1..256
          lex_out[base] = Lg[r][lbl] - lse;
        }
      }
    }
    __syncthreads();   // Lg reused by next tt
  }
}

// ---------------- forward DP over the lattice (log semiring) ----------------
__global__ __launch_bounds__(128) void dp_kernel(
    const float* __restrict__ blank, const float* __restrict__ lex,
    const int* __restrict__ num_frames, const int* __restrict__ num_labels,
    float* __restrict__ out)
{
  const int b = blockIdx.x;
  __shared__ float alpha[Ud + 1];
  __shared__ float anew[Ud + 1];
  const int u = threadIdx.x;
  if (u <= Ud) alpha[u] = (u == 0) ? 0.0f : NEGINF;
  __syncthreads();
  const int nf = num_frames[b];
  for (int t = 0; t < Td; ++t) {
    if (u <= Ud) {
      size_t base = ((size_t)(b * Td + t)) * UPAD;
      float stay = alpha[u] + blank[base + u];
      float emit = (u > 0) ? (alpha[u - 1] + lex[base + (u - 1)]) : NEGINF;
      float m = fmaxf(stay, emit);
      float d = fminf(stay, emit) - m;
      float nv = m + log1pf(__expf(d));
      anew[u] = (t < nf) ? nv : alpha[u];
    }
    __syncthreads();
    if (u <= Ud) alpha[u] = anew[u];
    __syncthreads();
  }
  if (threadIdx.x == 0) out[b] = -alpha[num_labels[b]];
}

// ---------------- launch ----------------
extern "C" void kernel_launch(void* const* d_in, const int* in_sizes, int n_in,
                              void* d_out, int out_size, void* d_ws, size_t ws_size,
                              hipStream_t stream) {
  const float* frames     = (const float*)d_in[0];
  const int*   num_frames = (const int*)  d_in[1];
  const int*   labels     = (const int*)  d_in[2];
  const int*   num_labels = (const int*)  d_in[3];
  const float* Wf         = (const float*)d_in[4];
  const float* E          = (const float*)d_in[5];
  const float* Wo         = (const float*)d_in[6];
  float* out = (float*)d_out;

  char* ws = (char*)d_ws;
  const size_t off_fproj  = 0;
  const size_t off_frbf   = off_fproj + (size_t)Bd*Td*Hd*4;
  const size_t off_wfT    = off_frbf  + (size_t)Bd*Td*Fd*2;
  const size_t off_woT    = off_wfT   + (size_t)Hd*Fd*2;
  const size_t off_cemb   = off_woT   + (size_t)NPW*Hd*2;
  const size_t off_blank  = off_cemb  + (size_t)Bd*UPAD*Hd*4;
  const size_t off_lex    = off_blank + (size_t)Bd*Td*UPAD*4;

  float*          fproj     = (float*)         (ws + off_fproj);
  unsigned short* frames_bf = (unsigned short*)(ws + off_frbf);
  unsigned short* wfT       = (unsigned short*)(ws + off_wfT);
  unsigned short* woT       = (unsigned short*)(ws + off_woT);
  float*          cemb      = (float*)         (ws + off_cemb);
  float*          blankb    = (float*)         (ws + off_blank);
  float*          lexb      = (float*)         (ws + off_lex);

  prep_frames_bf<<<dim3((Bd*Td*Fd + 255) / 256), dim3(256), 0, stream>>>(frames, frames_bf);
  prep_wfT      <<<dim3((Hd*Fd     + 255) / 256), dim3(256), 0, stream>>>(Wf, wfT);
  prep_woT      <<<dim3((NPW*Hd    + 255) / 256), dim3(256), 0, stream>>>(Wo, woT);
  prep_cemb     <<<dim3((Bd*UPAD*Hd+ 255) / 256), dim3(256), 0, stream>>>(E, labels, cemb);

  fproj_kernel<<<dim3(1024), dim3(128), 0, stream>>>(frames_bf, wfT, fproj);

  joint_kernel<<<dim3(Bd * (Td / 2) * 7), dim3(128), 0, stream>>>(
      fproj, cemb, woT, labels, blankb, lexb);

  dp_kernel<<<dim3(Bd), dim3(128), 0, stream>>>(blankb, lexb, num_frames, num_labels, out);
}
